// HarmonicCQT_13426067767882
// MI455X (gfx1250) — matrix operand order repaired and checked by reading, compiled, not run on hardware
//
#include <hip/hip_runtime.h>
#include <hip/hip_bf16.h>
#include <math.h>

typedef _Float16 v16h __attribute__((ext_vector_type(16)));
typedef _Float16 v8h  __attribute__((ext_vector_type(8)));
typedef float    v8f  __attribute__((ext_vector_type(8)));

// Problem constants (from the reference)
constexpr int BATCH  = 4;
constexpr int S      = 352768;          // samples per channel
constexpr int HOP    = 512;
constexpr int NBLK   = 689;             // S / HOP
constexpr int NBINS  = 168;
constexpr int NMAX   = 23014;           // longest CQT kernel
constexpr int PAD    = NMAX - HOP;      // 22502 left zero-pad
constexpr int KPAD   = 23040;           // NMAX rounded up to multiple of 32 (720 steps)
constexpr int TT     = 44;              // ceil(689/16) frame tiles
constexpr int NT     = 11;              // ceil(168/16) bin tiles
constexpr int BINPAD = 176;             // bins padded to 16*NT
constexpr int PL     = TT * 16 * HOP + KPAD;  // 383488 padded samples per batch (covers partial tile)

// ---- prep: f32 audio -> zero-padded f16 buffer [BATCH][PL] ----
__global__ __launch_bounds__(256) void cvt_audio_f16(const float* __restrict__ audio,
                                                     _Float16* __restrict__ out) {
  int i = blockIdx.x * 256 + threadIdx.x;
  if (i >= BATCH * PL) return;
  int b = i / PL;
  int q = i - b * PL;
  int s = q - PAD;
  float v = (s >= 0 && s < S) ? audio[(size_t)b * S + s] : 0.0f;
  out[i] = (_Float16)v;
}

// ---- prep: f32 kernels -> zero-padded f16 buffers [BINPAD][KPAD] ----
__global__ __launch_bounds__(256) void cvt_kern_f16(const float* __restrict__ kr,
                                                    const float* __restrict__ ki,
                                                    _Float16* __restrict__ okr,
                                                    _Float16* __restrict__ oki) {
  int i = blockIdx.x * 256 + threadIdx.x;
  if (i >= BINPAD * KPAD) return;
  int bin = i / KPAD;
  int n   = i - bin * KPAD;
  float vr = 0.0f, vi = 0.0f;
  if (bin < NBINS && n < NMAX) {
    size_t src = (size_t)bin * NMAX + n;
    vr = kr[src];
    vi = ki[src];
  }
  okr[i] = (_Float16)vr;
  oki[i] = (_Float16)vi;
}

// ---- main: one wave32 computes a 16x16 tile of (frames x bins), re+im via WMMA ----
__global__ __launch_bounds__(32) void cqt_wmma_f16(const _Float16* __restrict__ pa,
                                                   const _Float16* __restrict__ kr,
                                                   const _Float16* __restrict__ ki,
                                                   float* __restrict__ out) {
  int tile = blockIdx.x;
  int nb  = tile % NT;  int tmp = tile / NT;
  int tt  = tmp % TT;   int b   = tmp / TT;

  int lane = threadIdx.x;        // 0..31
  int lm   = lane & 15;          // row (A) / col (B) / N (D)
  int kh   = lane >> 4;          // K-half selector

  // A: frame row lm of this tile; per-lane contiguous K run [kh*16 .. kh*16+15]
  const _Float16* ar = pa + (size_t)b * PL + (size_t)(tt * 16 + lm) * HOP + kh * 16;
  // B: column = bin lm; per-lane contiguous K run [kh*16 .. kh*16+15]
  const _Float16* br = kr + (size_t)(nb * 16 + lm) * KPAD + kh * 16;
  const _Float16* bi = ki + (size_t)(nb * 16 + lm) * KPAD + kh * 16;

  v8f cre = {};
  v8f cim = {};

#pragma unroll 4
  for (int kk = 0; kk < KPAD; kk += 32) {
    union { v16h v; v8h h[2]; } A, Br, Bi;
    A.h[0]  = *(const v8h*)(ar + kk);        // K = kk + kh*16 .. +7
    A.h[1]  = *(const v8h*)(ar + kk + 8);    // K = kk + kh*16 + 8 .. +15
    Br.h[0] = *(const v8h*)(br + kk);        // K = kk + kh*16 .. +7
    Br.h[1] = *(const v8h*)(br + kk + 8);    // K = kk + kh*16 + 8 .. +15
    Bi.h[0] = *(const v8h*)(bi + kk);
    Bi.h[1] = *(const v8h*)(bi + kk + 8);
    cre = __builtin_amdgcn_wmma_f32_16x16x32_f16(false, A.v, false, Br.v,
                                                 (short)0, cre, false, false);
    cim = __builtin_amdgcn_wmma_f32_16x16x32_f16(false, A.v, false, Bi.v,
                                                 (short)0, cim, false, false);
  }

  // D layout: VGPR r, lanes 0-15 -> M=r, lanes 16-31 -> M=r+8; N = lane%16
  int bin = nb * 16 + lm;
  if (bin < NBINS) {
#pragma unroll
    for (int r = 0; r < 8; ++r) {
      int t = tt * 16 + r + kh * 8;
      if (t < NBLK) {
        float re = cre[r], im = cim[r];
        out[((size_t)b * NBLK + t) * NBINS + bin] = sqrtf(re * re + im * im);
      }
    }
  }
}

extern "C" void kernel_launch(void* const* d_in, const int* in_sizes, int n_in,
                              void* d_out, int out_size, void* d_ws, size_t ws_size,
                              hipStream_t stream) {
  const float* audio = (const float*)d_in[0];  // [4,1,352768] f32
  const float* kreal = (const float*)d_in[1];  // [168,23014] f32
  const float* kimag = (const float*)d_in[2];  // [168,23014] f32
  float* out = (float*)d_out;                  // [4,1,689,168] f32

  // workspace layout (all offsets multiples of 256 bytes)
  const size_t audioBytes = (size_t)BATCH * PL * sizeof(_Float16);      // 3,067,904
  const size_t kernBytes  = (size_t)BINPAD * KPAD * sizeof(_Float16);   // 8,110,080
  _Float16* aF16  = (_Float16*)d_ws;
  _Float16* krF16 = (_Float16*)((char*)d_ws + audioBytes);
  _Float16* kiF16 = (_Float16*)((char*)d_ws + audioBytes + kernBytes);

  {
    int total = BATCH * PL;
    cvt_audio_f16<<<(total + 255) / 256, 256, 0, stream>>>(audio, aF16);
  }
  {
    int total = BINPAD * KPAD;
    cvt_kern_f16<<<(total + 255) / 256, 256, 0, stream>>>(kreal, kimag, krF16, kiF16);
  }
  {
    int tiles = BATCH * TT * NT;  // 1936 wave32 tiles
    cqt_wmma_f16<<<tiles, 32, 0, stream>>>(aF16, krF16, kiF16, out);
  }
}
